// Self_Attention_39264591020603
// MI455X (gfx1250) — matrix operand order repaired and checked by reading, not compiled, not run on hardware
//
#include <hip/hip_runtime.h>
#include <hip/hip_bf16.h>

// ---------------------------------------------------------------------------
// Self-attention (SAGAN-style) for MI455X / gfx1250, wave32 + WMMA f16->f32.
//   B=4, C=256, C8=32, H=W=64, N=4096
//   out  = x + gamma * V @ softmax_col(Q^T K)          (4*256*4096 floats)
//   attn = softmax map, row-normalized over last axis  (4*4096*4096 floats)
// d_out = [out | attn] concatenated.
// Workspace: Qt[b][n][32] f16, Kt[b][n][32] f16, Vh[b][c][n] f16  (10 MB).
// gfx1250 paths: v_wmma_f32_16x16x32_f16, global_load_async_to_lds_b128
// (+ s_wait_asynccnt), global_prefetch_b8.
// ---------------------------------------------------------------------------

typedef _Float16 v16h __attribute__((ext_vector_type(16)));
typedef _Float16 h8   __attribute__((ext_vector_type(8)));
typedef float    v8f  __attribute__((ext_vector_type(8)));

#define NBATCH 4
#define CIN    256
#define CQK    32
#define NPIX   4096

static __device__ __forceinline__ v8f wmma_f16(v16h a, v16h b, v8f c) {
  // D = A(16x32 f16) * B(32x16 f16) + C(16x16 f32)
  return __builtin_amdgcn_wmma_f32_16x16x32_f16(false, a, false, b,
                                                (short)0, c, false, false);
}

static __device__ __forceinline__ v8f zero8() {
  v8f z = {0.f, 0.f, 0.f, 0.f, 0.f, 0.f, 0.f, 0.f};
  return z;
}

// Async copy 16B/lane global -> LDS (CDNA5 GLOBAL_LOAD_ASYNC_TO_LDS_B128,
// tracked by ASYNCcnt).  lds_off is the byte offset inside the block's LDS.
static __device__ __forceinline__ void async_copy_b128(unsigned lds_off,
                                                       const void* gptr) {
  asm volatile("global_load_async_to_lds_b128 %0, %1, off"
               :: "v"(lds_off), "v"(gptr)
               : "memory");
}
static __device__ __forceinline__ void wait_async0() {
  asm volatile("s_wait_asynccnt 0x0" ::: "memory");
}

// ---------------------------------------------------------------------------
// Kernel 1: fused QKV projection.
// Grid (N/32, B), block 256 (8 waves). Each block: 32 pixels of one batch.
// 40 (out-tile, n-tile) WMMA tiles per block -> 5 per wave (wave-uniform).
//   tiles p=0..3 : Q (2 o-tiles x 2 n-tiles)   D[M=n][N=o]
//   tiles p=4..7 : K                            D[M=n][N=o]
//   tiles p=8..39: V (16 o-tiles x 2 n-tiles)   D[M=c][N=n]
// x tile (32c x 32n fp32) staged via async global->LDS DMA each c-chunk.
// ---------------------------------------------------------------------------
__global__ __launch_bounds__(256) void proj_qkv_kernel(
    const float* __restrict__ x,
    const float* __restrict__ Wq, const float* __restrict__ bq,
    const float* __restrict__ Wk, const float* __restrict__ bk,
    const float* __restrict__ Wv, const float* __restrict__ bv,
    _Float16* __restrict__ Qt, _Float16* __restrict__ Kt,
    _Float16* __restrict__ Vh)
{
  __shared__ float ldsXf[32][36];     // [c-chunk][n] fp32, 144B row pitch

  const int b     = blockIdx.y;
  const int nbase = blockIdx.x * 32;
  const int tid   = threadIdx.x;
  const int wv    = __builtin_amdgcn_readfirstlane(tid >> 5);  // scalar wave id
  const int lane  = tid & 31;
  const int lo    = lane & 15;
  const int hi    = lane >> 4;

  // this thread's 16B slice of the 32x32 fp32 tile (4 consecutive floats)
  const int cc = tid >> 3;            // 0..31
  const int nn = (tid & 7) * 4;       // 0,4,...,28
  const unsigned lds_off =
      (unsigned)(unsigned long long)(&ldsXf[cc][nn]);

  v8f acc[5];
  #pragma unroll
  for (int s = 0; s < 5; ++s) acc[s] = zero8();

  for (int ch = 0; ch < 8; ++ch) {          // C = 8 chunks of 32
    const int c0 = ch * 32;
    __syncthreads();                        // protect previous-iter LDS reads
    async_copy_b128(lds_off,
                    &x[((size_t)b * CIN + c0 + cc) * NPIX + nbase + nn]);
    wait_async0();
    __syncthreads();

    #pragma unroll
    for (int s = 0; s < 5; ++s) {
      const int p = wv * 5 + s;             // SGPR -> scalar branches below
      v16h a, bm;
      if (p < 8) {
        // ---- Q/K tile: A = X^T (LDS, cvt f16), B = W^T (global, cvt f16)
        const int kind  = p >> 2;           // 0=Q 1=K
        const int q     = p & 3;
        const int otile = q >> 1;
        const int nt    = (q & 1) * 16;
        #pragma unroll
        for (int h = 0; h < 16; ++h) {
          int kc = 16 * hi + h;   // A-matrix K index
          a[h] = (_Float16)ldsXf[kc][nt + lo];
        }
        const float* wrow =
            (kind == 0 ? Wq : Wk) + (size_t)(otile * 16 + lo) * CIN + c0 + 16 * hi;
        const float4* wp = (const float4*)wrow;
        float4 w0 = wp[0], w1 = wp[1], w2 = wp[2], w3 = wp[3];
        bm[0]=(_Float16)w0.x;  bm[1]=(_Float16)w0.y;  bm[2]=(_Float16)w0.z;  bm[3]=(_Float16)w0.w;
        bm[4]=(_Float16)w1.x;  bm[5]=(_Float16)w1.y;  bm[6]=(_Float16)w1.z;  bm[7]=(_Float16)w1.w;
        bm[8]=(_Float16)w2.x;  bm[9]=(_Float16)w2.y;  bm[10]=(_Float16)w2.z; bm[11]=(_Float16)w2.w;
        bm[12]=(_Float16)w3.x; bm[13]=(_Float16)w3.y; bm[14]=(_Float16)w3.z; bm[15]=(_Float16)w3.w;
      } else {
        // ---- V tile: A = Wv rows (global, cvt f16), B = X (LDS, cvt f16)
        const int q     = p - 8;
        const int otile = q >> 1;
        const int nt    = (q & 1) * 16;
        const float* wrow = Wv + (size_t)(otile * 16 + lo) * CIN + c0;
        const float4* w0p = (const float4*)(wrow + 8 * hi);
        const float4* w1p = (const float4*)(wrow + 16 + 8 * hi);
        float4 u0 = w0p[0], u1 = w0p[1], u2 = w1p[0], u3 = w1p[1];
        a[0]=(_Float16)u0.x;  a[1]=(_Float16)u0.y;  a[2]=(_Float16)u0.z;  a[3]=(_Float16)u0.w;
        a[4]=(_Float16)u1.x;  a[5]=(_Float16)u1.y;  a[6]=(_Float16)u1.z;  a[7]=(_Float16)u1.w;
        a[8]=(_Float16)u2.x;  a[9]=(_Float16)u2.y;  a[10]=(_Float16)u2.z; a[11]=(_Float16)u2.w;
        a[12]=(_Float16)u3.x; a[13]=(_Float16)u3.y; a[14]=(_Float16)u3.z; a[15]=(_Float16)u3.w;
        #pragma unroll
        for (int h = 0; h < 16; ++h)
          bm[h] = (_Float16)ldsXf[(h < 8 ? h : h + 8) + 8 * hi][nt + lo];
      }
      acc[s] = wmma_f16(a, bm, acc[s]);
    }
  }

  // ---- epilogue: bias add + f16 store
  #pragma unroll
  for (int s = 0; s < 5; ++s) {
    const int p = wv * 5 + s;
    if (p < 8) {
      const int kind  = p >> 2;
      const int q     = p & 3;
      const int otile = q >> 1;
      const int nt    = (q & 1) * 16;
      const float bias = (kind == 0 ? bq : bk)[otile * 16 + lo];
      _Float16* dst = (kind == 0 ? Qt : Kt);
      #pragma unroll
      for (int r = 0; r < 8; ++r) {
        int nrow = nbase + nt + r + 8 * hi;
        dst[((size_t)b * NPIX + nrow) * CQK + otile * 16 + lo] =
            (_Float16)(acc[s][r] + bias);
      }
    } else {
      const int q     = p - 8;
      const int otile = q >> 1;
      const int nt    = (q & 1) * 16;
      #pragma unroll
      for (int r = 0; r < 8; ++r) {
        int crow = otile * 16 + r + 8 * hi;
        Vh[((size_t)b * CIN + crow) * NPIX + nbase + nt + lo] =
            (_Float16)(acc[s][r] + bv[crow]);
      }
    }
  }
}

// ---------------------------------------------------------------------------
// Kernel 2: attention map with two-pass (recompute) safe softmax.
// Grid (N/16 j-tiles, B), block 256 (8 waves).  Wave w covers i-chunks
// it = w, w+8, ... (32 chunks of 16 i each).  S tile D[M=j][N=i] = k_j . q_i.
// attn[b][j][i] = exp(S - max_i) / sum_i, written directly to d_out.
// ---------------------------------------------------------------------------
__global__ __launch_bounds__(256) void attn_softmax_kernel(
    const _Float16* __restrict__ Qt, const _Float16* __restrict__ Kt,
    float* __restrict__ attn)
{
  __shared__ float smM[8][16];
  __shared__ float smL[8][16];
  __shared__ float fM[16];
  __shared__ float fLinv[16];

  const int b    = blockIdx.y;
  const int jt   = blockIdx.x;
  const int tid  = threadIdx.x;
  const int wv   = __builtin_amdgcn_readfirstlane(tid >> 5);
  const int lane = tid & 31;
  const int lo   = lane & 15;
  const int hi   = lane >> 4;

  // A-operand (K tile, 16 j x 32 c) -- invariant over the i loop
  v16h a;
  {
    const _Float16* krow = Kt + ((size_t)b * NPIX + jt * 16 + lo) * CQK;
    h8 k0 = *(const h8*)(krow + 16 * hi);
    h8 k1 = *(const h8*)(krow + 16 * hi + 8);
    #pragma unroll
    for (int h = 0; h < 8; ++h) { a[h] = k0[h]; a[h + 8] = k1[h]; }
  }

  float m[8], l[8];
  #pragma unroll
  for (int r = 0; r < 8; ++r) { m[r] = -1e30f; l[r] = 0.f; }

  // ---- pass A: online (max,sum) per lane over this wave's i-chunks
  for (int it = wv; it < NPIX / 16; it += 8) {
    const int i0 = it * 16;
    v16h bm;
    const _Float16* qrow = Qt + ((size_t)b * NPIX + i0 + lo) * CQK + 16 * hi;
    h8 q0 = *(const h8*)qrow;
    h8 q1 = *(const h8*)(qrow + 8);
    #pragma unroll
    for (int h = 0; h < 8; ++h) { bm[h] = q0[h]; bm[h + 8] = q1[h]; }
    v8f dd = wmma_f16(a, bm, zero8());
    #pragma unroll
    for (int r = 0; r < 8; ++r) {
      float s  = dd[r];
      float mn = fmaxf(m[r], s);
      l[r] = l[r] * __expf(m[r] - mn) + __expf(s - mn);
      m[r] = mn;
    }
  }
  // butterfly merge across the 16 lanes that share each j row
  #pragma unroll
  for (int stp = 1; stp < 16; stp <<= 1) {
    #pragma unroll
    for (int r = 0; r < 8; ++r) {
      float mo = __shfl_xor(m[r], stp, 32);
      float lx = __shfl_xor(l[r], stp, 32);
      float mn = fmaxf(m[r], mo);
      l[r] = l[r] * __expf(m[r] - mn) + lx * __expf(mo - mn);
      m[r] = mn;
    }
  }
  if (lo == 0) {
    #pragma unroll
    for (int r = 0; r < 8; ++r) { smM[wv][r + 8 * hi] = m[r]; smL[wv][r + 8 * hi] = l[r]; }
  }
  __syncthreads();
  if (tid < 16) {                       // merge the 8 wave partials per j
    float M = -1e30f, L = 0.f;
    #pragma unroll
    for (int w = 0; w < 8; ++w) {
      float mw = smM[w][tid], lw = smL[w][tid];
      float mn = fmaxf(M, mw);
      L = L * __expf(M - mn) + lw * __expf(mw - mn);
      M = mn;
    }
    fM[tid] = M;
    fLinv[tid] = 1.0f / L;
  }
  __syncthreads();

  float Mf[8], Li[8];
  #pragma unroll
  for (int r = 0; r < 8; ++r) { Mf[r] = fM[r + 8 * hi]; Li[r] = fLinv[r + 8 * hi]; }

  // ---- pass B: recompute S (deterministic), write normalized probabilities
  float* out = attn + (size_t)b * NPIX * NPIX;
  for (int it = wv; it < NPIX / 16; it += 8) {
    const int i0 = it * 16;
    v16h bm;
    const _Float16* qrow = Qt + ((size_t)b * NPIX + i0 + lo) * CQK + 16 * hi;
    h8 q0 = *(const h8*)qrow;
    h8 q1 = *(const h8*)(qrow + 8);
    #pragma unroll
    for (int h = 0; h < 8; ++h) { bm[h] = q0[h]; bm[h + 8] = q1[h]; }
    v8f dd = wmma_f16(a, bm, zero8());
    #pragma unroll
    for (int r = 0; r < 8; ++r) {
      out[(size_t)(jt * 16 + r + 8 * hi) * NPIX + i0 + lo] =
          __expf(dd[r] - Mf[r]) * Li[r];
    }
  }
}

// ---------------------------------------------------------------------------
// Kernel 3: O = V @ A_T, out = x + gamma*O.
// Grid (N/16/8, C/16, B), block 256 (8 waves): wave w -> n-tile blockIdx.x*8+w.
// D[M=c][N=n]; A = Vh rows (f16), B = attn rows cvt f32->f16, K loop over m.
// ---------------------------------------------------------------------------
__global__ __launch_bounds__(256) void out_gemm_kernel(
    const float* __restrict__ x, const _Float16* __restrict__ Vh,
    const float* __restrict__ attn, const float* __restrict__ gamma,
    float* __restrict__ out)
{
  const int b     = blockIdx.z;
  const int cbase = blockIdx.y * 16;
  const int tid   = threadIdx.x;
  const int wv    = __builtin_amdgcn_readfirstlane(tid >> 5);
  const int lane  = tid & 31;
  const int lo    = lane & 15;
  const int hi    = lane >> 4;
  const int nbase = (blockIdx.x * 8 + wv) * 16;

  const _Float16* vrow = Vh + ((size_t)b * CIN + cbase + lo) * NPIX;
  const float*    arow = attn + (size_t)b * NPIX * NPIX + (size_t)(nbase + lo) * NPIX;

  v8f acc = zero8();
  for (int m0 = 0; m0 < NPIX; m0 += 32) {
    __builtin_prefetch(arow + ((m0 + 512) & (NPIX - 1)), 0, 0);  // global_prefetch_b8
    v16h a, bm;
    h8 va = *(const h8*)(vrow + m0 + 16 * hi);
    h8 vb = *(const h8*)(vrow + m0 + 16 * hi + 8);
    #pragma unroll
    for (int h = 0; h < 8; ++h) { a[h] = va[h]; a[h + 8] = vb[h]; }
    const float4* ap = (const float4*)(arow + m0 + 16 * hi);
    float4 f0 = ap[0], f1 = ap[1], f2 = ap[2], f3 = ap[3];
    bm[0]=(_Float16)f0.x;  bm[1]=(_Float16)f0.y;  bm[2]=(_Float16)f0.z;  bm[3]=(_Float16)f0.w;
    bm[4]=(_Float16)f1.x;  bm[5]=(_Float16)f1.y;  bm[6]=(_Float16)f1.z;  bm[7]=(_Float16)f1.w;
    bm[8]=(_Float16)f2.x;  bm[9]=(_Float16)f2.y;  bm[10]=(_Float16)f2.z; bm[11]=(_Float16)f2.w;
    bm[12]=(_Float16)f3.x; bm[13]=(_Float16)f3.y; bm[14]=(_Float16)f3.z; bm[15]=(_Float16)f3.w;
    acc = wmma_f16(a, bm, acc);
  }

  const float g = gamma[0];
  #pragma unroll
  for (int r = 0; r < 8; ++r) {
    size_t idx = ((size_t)b * CIN + cbase + r + 8 * hi) * NPIX + nbase + lo;
    out[idx] = x[idx] + g * acc[r];
  }
}

// ---------------------------------------------------------------------------
extern "C" void kernel_launch(void* const* d_in, const int* in_sizes, int n_in,
                              void* d_out, int out_size, void* d_ws, size_t ws_size,
                              hipStream_t stream) {
  const float* x     = (const float*)d_in[0];
  const float* Wq    = (const float*)d_in[1];
  const float* bq    = (const float*)d_in[2];
  const float* Wk    = (const float*)d_in[3];
  const float* bk    = (const float*)d_in[4];
  const float* Wv    = (const float*)d_in[5];
  const float* bv    = (const float*)d_in[6];
  const float* gamma = (const float*)d_in[7];

  float* out  = (float*)d_out;
  float* attn = out + (size_t)NBATCH * CIN * NPIX;   // 4,194,304 floats in

  _Float16* Qt = (_Float16*)d_ws;                          // 4*4096*32 f16
  _Float16* Kt = Qt + (size_t)NBATCH * NPIX * CQK;         // 4*4096*32 f16
  _Float16* Vh = Kt + (size_t)NBATCH * NPIX * CQK;         // 4*256*4096 f16

  proj_qkv_kernel<<<dim3(NPIX / 32, NBATCH), 256, 0, stream>>>(
      x, Wq, bq, Wk, bk, Wv, bv, Qt, Kt, Vh);
  attn_softmax_kernel<<<dim3(NPIX / 16, NBATCH), 256, 0, stream>>>(Qt, Kt, attn);
  out_gemm_kernel<<<dim3(NPIX / 16 / 8, CIN / 16, NBATCH), 256, 0, stream>>>(
      x, Vh, attn, gamma, out);
}
